// GAT_25778393711119
// MI455X (gfx1250) — matrix operand order repaired and checked by reading, not compiled, not run on hardware
//
#include <hip/hip_runtime.h>
#include <hip/hip_bf16.h>

// ---------------------------------------------------------------------------
// GAT (2-layer, PyG-style) for MI455X / gfx1250, wave32.
// GEMMs: v_wmma_f32_16x16x32_f16, one wave = 16 rows x (NT*16) cols,
// B pre-swizzled into fragment order (zero-padded), A read once per wave.
// Edge softmax/aggregation: float atomics (ordered-uint trick for max).
// ---------------------------------------------------------------------------

typedef __attribute__((ext_vector_type(16))) _Float16 v16h;
typedef __attribute__((ext_vector_type(8)))  _Float16 v8h;
typedef __attribute__((ext_vector_type(8)))  float    v8f;

#define NEG_SLOPE 0.2f

static __device__ __forceinline__ unsigned f2ord(float f) {
    unsigned u = __float_as_uint(f);
    return (u & 0x80000000u) ? ~u : (u | 0x80000000u);
}
static __device__ __forceinline__ float ord2f(unsigned u) {
    return __uint_as_float((u & 0x80000000u) ? (u ^ 0x80000000u) : ~u);
}
static __device__ __forceinline__ void edge_sd(const int* ei, int E, int e,
                                               int& s, int& d) {
    if (e < E) { s = ei[e]; d = ei[E + e]; }
    else       { s = e - E; d = e - E; }        // appended self-loops
}
static __device__ __forceinline__ v16h cat16(v8h lo, v8h hi) {
    return __builtin_shufflevector(lo, hi, 0, 1, 2, 3, 4, 5, 6, 7,
                                           8, 9, 10, 11, 12, 13, 14, 15);
}

// -------------------------------- casts ------------------------------------
__global__ void k_f32_to_f16(const float* __restrict__ s,
                             _Float16* __restrict__ d, long n) {
    long i = (long)blockIdx.x * blockDim.x + threadIdx.x;
    if (i < n) d[i] = (_Float16)s[i];
}

// relu(agg + bias) -> f16 (input to layer-2 GEMM)
__global__ void k_relu_bias_cast(const float* __restrict__ agg,
                                 const float* __restrict__ bias,
                                 _Float16* __restrict__ d, long n, int HC) {
    long i = (long)blockIdx.x * blockDim.x + threadIdx.x;
    if (i >= n) return;
    float v = agg[i] + bias[(int)(i % HC)];
    d[i] = (_Float16)fmaxf(v, 0.0f);
}

__global__ void k_fill_f32(float* __restrict__ p, float v, long n) {
    long i = (long)blockIdx.x * blockDim.x + threadIdx.x;
    if (i < n) p[i] = v;
}
__global__ void k_fill_ord(unsigned* __restrict__ p, float v, long n) {
    long i = (long)blockIdx.x * blockDim.x + threadIdx.x;
    if (i < n) p[i] = f2ord(v);
}

// ------------------- pack B (f32 KxNreal) into WMMA fragments --------------
// Bs layout: [kb][nt][lane][16 halves] so each lane's fragment is 32
// contiguous bytes. Zero-padded for col >= Nreal.
// Fragment order (contiguous K, matching the A loads in k_gemm_wmma): lane<16 owns col=lane, K=kb*32+0..15;
// lane>=16 owns col=lane-16, K=kb*32+16..31; element i <-> K offset i.
__global__ void k_pack_B(const float* __restrict__ B, _Float16* __restrict__ Bs,
                         int K, int Nreal, int NT) {
    int t = blockIdx.x * blockDim.x + threadIdx.x;
    int KB = K >> 5;
    if (t >= KB * NT * 32) return;
    int lane = t & 31;
    int nt   = (t >> 5) % NT;
    int kb   = (t >> 5) / NT;
    int hi = lane >> 4, l15 = lane & 15;
    int col = nt * 16 + l15;
    int kbase = kb * 32 + hi * 16;
    _Float16* o = Bs + (size_t)t * 16;
#pragma unroll
    for (int i = 0; i < 16; ++i) {
        float v = (col < Nreal) ? B[(size_t)(kbase + i) * Nreal + col] : 0.0f;
        o[i] = (_Float16)v;
    }
}

// ------------------------------ WMMA GEMM ----------------------------------
// C[M x (NT*16)] (row stride ldc) = A[M x K] * B (pre-swizzled, NT n-tiles).
// One wave computes a 16 x NT*16 strip; 8 waves per block along M.
template <int NT>
__global__ void k_gemm_wmma(const _Float16* __restrict__ A,
                            const _Float16* __restrict__ Bs,
                            float* __restrict__ C,
                            int M, int K, int ldc) {
    const int lane = threadIdx.x & 31;
    const int wave = threadIdx.x >> 5;
    const int mt = blockIdx.x * (blockDim.x >> 5) + wave;  // M tile
    if (mt * 16 >= M) return;                              // wave-uniform

    const int hi  = lane >> 4;        // 0: lanes 0-15, 1: lanes 16-31
    const int l15 = lane & 15;
    const int row = mt * 16 + l15;    // A row owned by this lane

    const _Float16* arow = A + (size_t)row * K + hi * 16;  // 16B aligned
    const _Float16* bp   = Bs + (size_t)lane * 16;         // 32B aligned

    v8f acc[NT];
#pragma unroll
    for (int nt = 0; nt < NT; ++nt) acc[nt] = (v8f){};

    for (int k0 = 0; k0 < K; k0 += 32) {
        v8h alo = *(const v8h*)(arow);        // K = k0+akofs+0..7
        v8h ahi = *(const v8h*)(arow + 8);    // K = k0+akofs+8..15
        v16h a = cat16(alo, ahi);
#pragma unroll
        for (int nt = 0; nt < NT; ++nt) {
            const _Float16* bq = bp + (size_t)nt * (32 * 16);
            v16h b = cat16(*(const v8h*)(bq), *(const v8h*)(bq + 8));
            acc[nt] = __builtin_amdgcn_wmma_f32_16x16x32_f16(
                false, a, false, b, (short)0, acc[nt], false, false);
        }
        arow += 32;
        bp   += (size_t)NT * (32 * 16);
    }
#pragma unroll
    for (int nt = 0; nt < NT; ++nt) {
#pragma unroll
        for (int v = 0; v < 8; ++v) {          // D: row = v + 8*hi, col = l15
            int r = mt * 16 + v + hi * 8;
            int c = nt * 16 + l15;
            C[(size_t)r * ldc + c] = acc[nt][v];
        }
    }
}

// --------------------------- attention alphas ------------------------------
// alpha_{src,dst}[n,h] = dot(h[n, h*C : h*C+C], att_{src,dst}[h,:])
__global__ void k_alpha(const float* __restrict__ h, int ld,
                        const float* __restrict__ att_s,
                        const float* __restrict__ att_d,
                        float* __restrict__ as, float* __restrict__ ad,
                        int n, int H, int C) {
    int t = blockIdx.x * blockDim.x + threadIdx.x;
    if (t >= n * H) return;
    int node = t / H, hd = t % H;
    const float* hp = h + (size_t)node * ld + (size_t)hd * C;
    const float* sp = att_s + hd * C;
    const float* dp = att_d + hd * C;
    float a = 0.f, b = 0.f;
    for (int c = 0; c < C; ++c) { a += hp[c] * sp[c]; b += hp[c] * dp[c]; }
    as[t] = a; ad[t] = b;
}

// ------------------------- edge softmax (3 passes) -------------------------
__global__ void k_edge_max(const int* __restrict__ ei, int E, int ET,
                           const float* __restrict__ as,
                           const float* __restrict__ ad,
                           unsigned* __restrict__ m, int H) {
    int e = blockIdx.x * blockDim.x + threadIdx.x;
    if (e >= ET) return;
    int s, d; edge_sd(ei, E, e, s, d);
    for (int h = 0; h < H; ++h) {
        float el = as[s * H + h] + ad[d * H + h];
        el = el > 0.f ? el : el * NEG_SLOPE;
        atomicMax(&m[d * H + h], f2ord(el));
    }
}

__global__ void k_edge_w(const int* __restrict__ ei, int E, int ET,
                         const float* __restrict__ as,
                         const float* __restrict__ ad,
                         const unsigned* __restrict__ m,
                         float* __restrict__ w, float* __restrict__ den,
                         int H) {
    int e = blockIdx.x * blockDim.x + threadIdx.x;
    if (e >= ET) return;
    int s, d; edge_sd(ei, E, e, s, d);
    for (int h = 0; h < H; ++h) {
        float el = as[s * H + h] + ad[d * H + h];
        el = el > 0.f ? el : el * NEG_SLOPE;
        float ww = __expf(el - ord2f(m[d * H + h]));
        w[(long)e * H + h] = ww;
        atomicAdd(&den[d * H + h], ww);
    }
}

// agg[dst, c] += h[src, c] * alpha[e, head(c)]  -- one thread per (edge, c)
__global__ void k_edge_aggr(const int* __restrict__ ei, int E, int ET,
                            const float* __restrict__ h, int ld,
                            const float* __restrict__ w,
                            const float* __restrict__ den,
                            float* __restrict__ agg, int H, int C) {
    long t = (long)blockIdx.x * blockDim.x + threadIdx.x;
    int HC = H * C;
    long tot = (long)ET * HC;
    if (t >= tot) return;
    int e = (int)(t / HC);
    int c = (int)(t % HC);
    int hd = c / C;
    int s, d; edge_sd(ei, E, e, s, d);
    float alpha = w[(long)e * H + hd] / (den[d * H + hd] + 1e-16f);
    atomicAdd(&agg[(long)d * HC + c], h[(long)s * ld + c] * alpha);
}

// ------------------------- final bias + log_softmax ------------------------
// one wave (32 lanes) per node, C = 41 classes (lane handles c and c+32)
__global__ void k_logsoftmax(const float* __restrict__ agg,
                             const float* __restrict__ bias,
                             float* __restrict__ out, int n, int C) {
    int node = (int)((blockIdx.x * (long)blockDim.x + threadIdx.x) >> 5);
    int lane = threadIdx.x & 31;
    if (node >= n) return;                                 // wave-uniform
    const float* row = agg + (size_t)node * C;
    float v0 = (lane      < C) ? row[lane]      + bias[lane]      : -3.0e38f;
    float v1 = (lane + 32 < C) ? row[lane + 32] + bias[lane + 32] : -3.0e38f;
    float mx = fmaxf(v0, v1);
    for (int o = 16; o; o >>= 1) mx = fmaxf(mx, __shfl_xor(mx, o));
    float s = ((lane      < C) ? __expf(v0 - mx) : 0.f)
            + ((lane + 32 < C) ? __expf(v1 - mx) : 0.f);
    for (int o = 16; o; o >>= 1) s += __shfl_xor(s, o);
    float lse = mx + __logf(s);
    float* orow = out + (size_t)node * C;
    if (lane      < C) orow[lane]      = v0 - lse;
    if (lane + 32 < C) orow[lane + 32] = v1 - lse;
}

// ---------------------------------------------------------------------------
extern "C" void kernel_launch(void* const* d_in, const int* in_sizes, int n_in,
                              void* d_out, int out_size, void* d_ws,
                              size_t ws_size, hipStream_t stream) {
    const int N = 100000, E = 1600000, ET = E + N;
    const int INC = 256, H1 = 4, C1 = 32, HC1 = 128, C2 = 41;
    const int NT1 = HC1 / 16;          // 8 n-tiles, layer-1 GEMM
    const int NT2 = (C2 + 15) / 16;    // 3 n-tiles, layer-2 GEMM
    const int LD2 = NT2 * 16;          // padded row stride of h2 (=48)

    const float* x     = (const float*)d_in[0];
    const int*   ei    = (const int*)  d_in[1];
    const float* W1    = (const float*)d_in[2];
    const float* atts1 = (const float*)d_in[3];
    const float* attd1 = (const float*)d_in[4];
    const float* b1    = (const float*)d_in[5];
    const float* W2    = (const float*)d_in[6];
    const float* atts2 = (const float*)d_in[7];
    const float* attd2 = (const float*)d_in[8];
    const float* b2    = (const float*)d_in[9];
    float* out = (float*)d_out;

    // ---- workspace carve-up (256B aligned) ----
    char* wsb = (char*)d_ws;
    size_t off = 0;
    auto take = [&](size_t bytes) -> char* {
        char* p = wsb + off;
        off += (bytes + 255) & ~(size_t)255;
        return p;
    };
    _Float16* xh  = (_Float16*)take((size_t)N * INC * 2);            // 51.2 MB
    _Float16* Bs1 = (_Float16*)take((size_t)(INC / 32) * NT1 * 32 * 16 * 2);
    _Float16* Bs2 = (_Float16*)take((size_t)(HC1 / 32) * NT2 * 32 * 16 * 2);
    float*    h1  = (float*)   take((size_t)N * HC1 * 4);            // 51.2 MB
    float*    as1 = (float*)   take((size_t)N * H1 * 4);
    float*    ad1 = (float*)   take((size_t)N * H1 * 4);
    unsigned* m1  = (unsigned*)take((size_t)N * H1 * 4);
    float*    den1= (float*)   take((size_t)N * H1 * 4);
    float*    w1  = (float*)   take((size_t)ET * H1 * 4);            // 27.2 MB
    float*    as2 = (float*)   take((size_t)N * 4);
    float*    ad2 = (float*)   take((size_t)N * 4);
    unsigned* m2  = (unsigned*)take((size_t)N * 4);
    float*    den2= (float*)   take((size_t)N * 4);
    float*    w2  = (float*)   take((size_t)ET * 4);                 // 6.8 MB
    // aliases over dead regions (pipeline is strictly sequential on stream):
    float*    agg1 = (float*)xh;      // x_f16 dead after GEMM1
    _Float16* h1r  = (_Float16*)h1;   // h1 dead after layer-1 aggregation
    float*    h2   = w1;              // w1 dead after layer-1 aggregation (N*48*4 = 19.2MB <= 27.2MB)
    float*    agg2 = (float*)xh;      // agg1 dead after relu-cast

    auto cdiv = [](long a, long b) { return (unsigned)((a + b - 1) / b); };

    // 1) cast x, pack weights into WMMA fragment order (zero-padded)
    k_f32_to_f16<<<cdiv((long)N * INC, 256), 256, 0, stream>>>(x, xh, (long)N * INC);
    k_pack_B<<<cdiv((long)(INC / 32) * NT1 * 32, 256), 256, 0, stream>>>(W1, Bs1, INC, HC1, NT1);
    k_pack_B<<<cdiv((long)(HC1 / 32) * NT2 * 32, 256), 256, 0, stream>>>(W2, Bs2, HC1, C2, NT2);

    // 2) layer-1 GEMM: h1 = x @ W1   (6250 m-tiles, 8 waves/block)
    k_gemm_wmma<NT1><<<cdiv(N / 16, 8), 256, 0, stream>>>(xh, Bs1, h1, N, INC, HC1);

    // 3) layer-1 attention
    k_fill_f32<<<cdiv((long)N * HC1, 256), 256, 0, stream>>>(agg1, 0.f, (long)N * HC1);
    k_alpha<<<cdiv((long)N * H1, 256), 256, 0, stream>>>(h1, HC1, atts1, attd1, as1, ad1, N, H1, C1);
    k_fill_ord<<<cdiv((long)N * H1, 256), 256, 0, stream>>>(m1, -3.0e38f, (long)N * H1);
    k_fill_f32<<<cdiv((long)N * H1, 256), 256, 0, stream>>>(den1, 0.f, (long)N * H1);
    k_edge_max<<<cdiv(ET, 256), 256, 0, stream>>>(ei, E, ET, as1, ad1, m1, H1);
    k_edge_w  <<<cdiv(ET, 256), 256, 0, stream>>>(ei, E, ET, as1, ad1, m1, w1, den1, H1);
    k_edge_aggr<<<cdiv((long)ET * HC1, 256), 256, 0, stream>>>(ei, E, ET, h1, HC1, w1, den1, agg1, H1, C1);

    // 4) relu(agg1 + b1) -> f16, then layer-2 GEMM: h2 = h1r @ W2 (ldc=48)
    k_relu_bias_cast<<<cdiv((long)N * HC1, 256), 256, 0, stream>>>(agg1, b1, h1r, (long)N * HC1, HC1);
    k_gemm_wmma<NT2><<<cdiv(N / 16, 8), 256, 0, stream>>>(h1r, Bs2, h2, N, HC1, LD2);

    // 5) layer-2 attention (1 head, 41 channels; mean over 1 head == identity)
    k_alpha<<<cdiv(N, 256), 256, 0, stream>>>(h2, LD2, atts2, attd2, as2, ad2, N, 1, C2);
    k_fill_ord<<<cdiv(N, 256), 256, 0, stream>>>(m2, -3.0e38f, N);
    k_fill_f32<<<cdiv(N, 256), 256, 0, stream>>>(den2, 0.f, N);
    k_fill_f32<<<cdiv((long)N * C2, 256), 256, 0, stream>>>(agg2, 0.f, (long)N * C2);
    k_edge_max<<<cdiv(ET, 256), 256, 0, stream>>>(ei, E, ET, as2, ad2, m2, 1);
    k_edge_w  <<<cdiv(ET, 256), 256, 0, stream>>>(ei, E, ET, as2, ad2, m2, w2, den2, 1);
    k_edge_aggr<<<cdiv((long)ET * C2, 256), 256, 0, stream>>>(ei, E, ET, h2, LD2, w2, den2, agg2, 1, C2);

    // 6) out = log_softmax(agg2 + b2)
    k_logsoftmax<<<cdiv((long)N * 32, 256), 256, 0, stream>>>(agg2, b2, out, N, C2);
}
